// TMMixer_42356967473848
// MI455X (gfx1250) — matrix operand order repaired and checked by reading, not compiled, not run on hardware
//
#include <hip/hip_runtime.h>
#include <hip/hip_bf16.h>

// MI455X / gfx1250 implementation of the TMMixer block.
// Matmuls on v_wmma_f32_16x16x32_f16 (wave32); weight tiles staged into LDS
// via gfx1250 async global->LDS copies (ASYNCcnt / s_wait_asynccnt).

#define DEV __device__ __forceinline__

typedef __attribute__((ext_vector_type(16))) _Float16 v16h;
typedef __attribute__((ext_vector_type(8)))  _Float16 v8h;
typedef __attribute__((ext_vector_type(8)))  float    v8f;

static constexpr int BATCH = 64;
static constexpr int LT  = 64;     // query tokens
static constexpr int LS  = 256;    // kv tokens
static constexpr int TOK = LT + LS;
static constexpr int DIM = 768;
static constexpr int HEADS = 12;
static constexpr int HD  = 64;     // head dim
static constexpr int ZROWS = 2 * BATCH * LT;    // 8192
static constexpr int XROWS = 2 * BATCH * LS;    // 32768
static constexpr int YROWS = 2 * BATCH * TOK;   // 40960

DEV int lane_id() { return (int)(threadIdx.x & 31u); }

DEV v16h pack16(v8h lo, v8h hi) {
  v16h f;
#pragma unroll
  for (int i = 0; i < 8; ++i) { f[i] = lo[i]; f[i + 8] = hi[i]; }
  return f;
}

// ---------- WMMA fragment helpers (layouts per CDNA5 ISA 7.12.2) ----------

// A fragment (16x32 f16): lane holds row M=lane%16; K pattern
//   half h=lane/16: k = {16h..16h+15} contiguous (matches frag_b).
DEV v16h frag_a(const _Float16* p, int ld, int row, int col) {
  int l = lane_id();
  const _Float16* base = p + (long)(row + (l & 15)) * ld + col + 16 * (l >> 4);
  return pack16(*(const v8h*)base, *(const v8h*)(base + 8));
}

// Same pattern, but source is f32 (e.g. attention probs in LDS) -> convert.
DEV v16h frag_a_f32(const float* p, int ld, int row, int col) {
  int l = lane_id();
  const float* base = p + (long)(row + (l & 15)) * ld + col + 16 * (l >> 4);
  v16h f;
#pragma unroll
  for (int i = 0; i < 8; ++i) {
    f[i]     = (_Float16)base[i];
    f[i + 8] = (_Float16)base[8 + i];
  }
  return f;
}

// B fragment for B^T stored row-major (N x K): lane holds row n=lane%16,
// 16 contiguous k values starting at 16*(lane/16)  (same k order as frag_a).
DEV v16h frag_b(const _Float16* p, int ld, int nrow, int col) {
  int l = lane_id();
  const _Float16* base = p + (long)(nrow + (l & 15)) * ld + col + 16 * (l >> 4);
  return pack16(*(const v8h*)base, *(const v8h*)(base + 8));
}

DEV v8f wmma_f16(v16h a, v16h b, v8f c) {
  return __builtin_amdgcn_wmma_f32_16x16x32_f16(false, a, false, b, (short)0, c,
                                                false, false);
}

DEV float gelu_tanh(float x) {  // jax.nn.gelu (approximate=True)
  float x3 = x * x * x;
  return 0.5f * x * (1.0f + tanhf(0.7978845608028654f * (x + 0.044715f * x3)));
}

// gfx1250 async global->LDS copy of 32 contiguous bytes (2 x b128).
// INST_OFFSET applies to both the LDS and the global address (ISA 10.x).
DEV void async_cp32(unsigned lds_byte_off, const void* gaddr) {
  asm volatile(
      "global_load_async_to_lds_b128 %0, %1, off\n\t"
      "global_load_async_to_lds_b128 %0, %1, off offset:16"
      :: "v"(lds_byte_off), "v"((unsigned long long)(size_t)gaddr) : "memory");
}
DEV void wait_async0() { asm volatile("s_wait_asynccnt 0x0" ::: "memory"); }

// ---------- weight prep: dst(N x K) f16 = transpose(src(K x N) f32) ----------
__global__ void wprep_kernel(const float* __restrict__ src, _Float16* __restrict__ dst,
                             int K, int N) {
  int i = blockIdx.x * 256 + threadIdx.x;   // i = n*K + k (coalesced dst)
  if (i >= K * N) return;
  int n = i / K, k = i % K;
  dst[i] = (_Float16)src[(long)k * N + n];
}

// ---------- layernorm ----------
__device__ void ln_token(const float* src, const float* w, const float* b,
                         _Float16* dst) {
  __shared__ float red[256];
  int tid = threadIdx.x;
  float v0 = src[tid], v1 = src[tid + 256], v2 = src[tid + 512];
  red[tid] = v0 + v1 + v2;
  __syncthreads();
  for (int o = 128; o > 0; o >>= 1) { if (tid < o) red[tid] += red[tid + o]; __syncthreads(); }
  float mu = red[0] * (1.0f / 768.0f);
  __syncthreads();
  float d0 = v0 - mu, d1 = v1 - mu, d2 = v2 - mu;
  red[tid] = d0 * d0 + d1 * d1 + d2 * d2;
  __syncthreads();
  for (int o = 128; o > 0; o >>= 1) { if (tid < o) red[tid] += red[tid + o]; __syncthreads(); }
  float rstd = rsqrtf(red[0] * (1.0f / 768.0f) + 1e-5f);
  __syncthreads();
  dst[tid]       = (_Float16)(d0 * rstd * w[tid]       + b[tid]);
  dst[tid + 256] = (_Float16)(d1 * rstd * w[tid + 256] + b[tid + 256]);
  dst[tid + 512] = (_Float16)(d2 * rstd * w[tid + 512] + b[tid + 512]);
}

// ln1 over both inputs, scattering z tokens and x tokens to separate buffers.
__global__ void ln1_kernel(const float* __restrict__ x0, const float* __restrict__ x1,
                           const float* __restrict__ w, const float* __restrict__ b,
                           _Float16* __restrict__ lnz, _Float16* __restrict__ lnx) {
  int t = blockIdx.x;                  // 0 .. 40959 : [m][b][pos]
  int m = t / (BATCH * TOK);
  int r = t % (BATCH * TOK);
  int bi = r / TOK, pos = r % TOK;
  const float* src = (m == 0 ? x0 : x1) + (long)(bi * TOK + pos) * DIM;
  _Float16* dst = (pos < LT)
      ? lnz + (long)((m * BATCH + bi) * LT + pos) * DIM
      : lnx + (long)((m * BATCH + bi) * LS + (pos - LT)) * DIM;
  ln_token(src, w, b, dst);
}

__global__ void ln2_kernel(const float* __restrict__ y, const float* __restrict__ w,
                           const float* __restrict__ b, _Float16* __restrict__ out) {
  long t = blockIdx.x;
  ln_token(y + t * DIM, w, b, out + t * DIM);
}

// ---------- GEMM epilogues ----------
enum { GM_F32 = 0, GM_F16 = 1, GM_GELU_F16 = 2, GM_TRANS_F16 = 3, GM_RESID_F32 = 4 };

template <int MODE>
DEV void store_tile(const v8f& acc, int m0, int c0, int N,
                    float* outf, _Float16* outh, const float* resid, int group_rows) {
  int l = lane_id(), hl = l >> 4;
  int c = c0 + (l & 15);
#pragma unroll
  for (int v = 0; v < 8; ++v) {
    int r = m0 + 8 * hl + v;
    float x = acc[v];
    if (MODE == GM_F32)            outf[(long)r * N + c] = x;
    else if (MODE == GM_F16)       outh[(long)r * N + c] = (_Float16)x;
    else if (MODE == GM_GELU_F16)  outh[(long)r * N + c] = (_Float16)gelu_tanh(x);
    else if (MODE == GM_TRANS_F16) {
      int g = r / group_rows, rl = r % group_rows;       // store C^T per group
      outh[((long)g * N + c) * group_rows + rl] = (_Float16)x;
    } else {                       // GM_RESID_F32
      outf[(long)r * N + c] = x + resid[(long)r * N + c];
    }
  }
}

// ---------- wide GEMM (N multiple of 128): C = A(MxK) * W, Bt = W^T (NxK) ----
// 8 waves, block 64Mx128N. Per k-step the 128x32 B tile is staged into LDS
// once per block via async global->LDS b128 copies (double buffered), so the
// copy for step s+1 overlaps the 4 WMMAs of step s. A is register-pipelined.
template <int MODE>
__global__ void gemm_wide_kernel(const _Float16* __restrict__ A, const _Float16* __restrict__ Bt,
                                 int M, int N, int K,
                                 float* __restrict__ outf, _Float16* __restrict__ outh,
                                 const float* __restrict__ resid, int group_rows) {
  extern __shared__ char smem[];
  _Float16* buf0 = (_Float16*)smem;                    // [128][32] f16
  _Float16* buf1 = (_Float16*)(smem + 128 * 32 * 2);   // [128][32] f16
  int tid = (int)threadIdx.x, wave = tid >> 5, l = tid & 31;
  int m0 = blockIdx.x * 64 + 16 * (wave & 3);
  int nb_blk = blockIdx.y * 128;
  int nw = 64 * (wave >> 2);

  // cooperative copy: each thread owns 32B of one row of the 128x32 tile
  int brow = tid >> 1;
  int bcol = (tid & 1) * 16;                           // elements
  const _Float16* gsrc = Bt + (long)(nb_blk + brow) * K + bcol;
  unsigned lds0 = (unsigned)(size_t)(buf0 + brow * 32 + bcol);
  unsigned lds1 = (unsigned)(size_t)(buf1 + brow * 32 + bcol);

  const int steps = K >> 5;
  async_cp32(lds0, gsrc);                              // stage k-step 0

  const _Float16* pa = A + (long)(m0 + (l & 15)) * K + 16 * (l >> 4);
  v16h a = pack16(*(const v8h*)pa, *(const v8h*)(pa + 8));
  v8f acc0 = {}, acc1 = {}, acc2 = {}, acc3 = {};

  for (int s = 0; s < steps; ++s) {
    wait_async0();            // own copies into buf[s&1] complete
    __syncthreads();          // whole tile valid; prior reads of other buf done
    if (s + 1 < steps)
      async_cp32((s + 1) & 1 ? lds1 : lds0, gsrc + (s + 1) * 32);
    const _Float16* bb = (s & 1) ? buf1 : buf0;
    v16h an = a;
    if (s + 1 < steps) {
      pa += 32;
      an = pack16(*(const v8h*)pa, *(const v8h*)(pa + 8));
    }
    v16h b0 = frag_b(bb, 32, nw,      0);
    v16h b1 = frag_b(bb, 32, nw + 16, 0);
    v16h b2 = frag_b(bb, 32, nw + 32, 0);
    v16h b3 = frag_b(bb, 32, nw + 48, 0);
    acc0 = wmma_f16(a, b0, acc0);
    acc1 = wmma_f16(a, b1, acc1);
    acc2 = wmma_f16(a, b2, acc2);
    acc3 = wmma_f16(a, b3, acc3);
    a = an;
  }
  int c0 = nb_blk + nw;
  store_tile<MODE>(acc0, m0, c0,      N, outf, outh, resid, group_rows);
  store_tile<MODE>(acc1, m0, c0 + 16, N, outf, outh, resid, group_rows);
  store_tile<MODE>(acc2, m0, c0 + 32, N, outf, outh, resid, group_rows);
  store_tile<MODE>(acc3, m0, c0 + 48, N, outf, outh, resid, group_rows);
}

// ---------- narrow GEMM (N == 64): block 128Mx64N, direct register pipeline --
template <int MODE>
__global__ void gemm_narrow_kernel(const _Float16* __restrict__ A, const _Float16* __restrict__ Bt,
                                   int M, int N, int K,
                                   float* __restrict__ outf, _Float16* __restrict__ outh,
                                   const float* __restrict__ resid, int group_rows) {
  int wave = (int)(threadIdx.x >> 5);
  int l = lane_id();
  int m0 = blockIdx.x * 128 + 16 * wave;

  const _Float16* pa = A  + (long)(m0 + (l & 15)) * K + 16 * (l >> 4);
  const _Float16* pb = Bt + (long)(l & 15) * K + 16 * (l >> 4);
  const long bs = (long)16 * K;

  v8f acc0 = {}, acc1 = {}, acc2 = {}, acc3 = {};
  v16h a  = pack16(*(const v8h*)(pa),            *(const v8h*)(pa + 8));
  v16h b0 = pack16(*(const v8h*)(pb),            *(const v8h*)(pb + 8));
  v16h b1 = pack16(*(const v8h*)(pb + bs),       *(const v8h*)(pb + bs + 8));
  v16h b2 = pack16(*(const v8h*)(pb + 2 * bs),   *(const v8h*)(pb + 2 * bs + 8));
  v16h b3 = pack16(*(const v8h*)(pb + 3 * bs),   *(const v8h*)(pb + 3 * bs + 8));

  const int steps = K >> 5;
  for (int s = 1; s < steps; ++s) {
    pa += 32; pb += 32;
    v16h an  = pack16(*(const v8h*)(pa),          *(const v8h*)(pa + 8));
    v16h bn0 = pack16(*(const v8h*)(pb),          *(const v8h*)(pb + 8));
    v16h bn1 = pack16(*(const v8h*)(pb + bs),     *(const v8h*)(pb + bs + 8));
    v16h bn2 = pack16(*(const v8h*)(pb + 2 * bs), *(const v8h*)(pb + 2 * bs + 8));
    v16h bn3 = pack16(*(const v8h*)(pb + 3 * bs), *(const v8h*)(pb + 3 * bs + 8));
    acc0 = wmma_f16(a, b0, acc0);
    acc1 = wmma_f16(a, b1, acc1);
    acc2 = wmma_f16(a, b2, acc2);
    acc3 = wmma_f16(a, b3, acc3);
    a = an; b0 = bn0; b1 = bn1; b2 = bn2; b3 = bn3;
  }
  acc0 = wmma_f16(a, b0, acc0);
  acc1 = wmma_f16(a, b1, acc1);
  acc2 = wmma_f16(a, b2, acc2);
  acc3 = wmma_f16(a, b3, acc3);

  store_tile<MODE>(acc0, m0, 0,  N, outf, outh, resid, group_rows);
  store_tile<MODE>(acc1, m0, 16, N, outf, outh, resid, group_rows);
  store_tile<MODE>(acc2, m0, 32, N, outf, outh, resid, group_rows);
  store_tile<MODE>(acc3, m0, 48, N, outf, outh, resid, group_rows);
}

// ---------- attention: one workgroup per (b, modality); loop over 12 heads ----
// dyn LDS: logits 64x256 f32 (64KB) + affinity 256x64 f32 (64KB) = 128KB.
__global__ void attn_kernel(const _Float16* __restrict__ Q, const _Float16* __restrict__ Kb,
                            const _Float16* __restrict__ Vt, _Float16* __restrict__ AO,
                            float* __restrict__ AFF) {
  extern __shared__ char smem[];
  float* logits = (float*)smem;                  // [64][256]
  float* aff    = (float*)(smem + 64 * 256 * 4); // [256][64]
  int b = blockIdx.x, m = blockIdx.y;
  int tid = threadIdx.x, wave = tid >> 5, l = tid & 31;
  const _Float16* Qp  = Q  + (long)((m * BATCH + b) * LT) * DIM;
  const _Float16* Kp  = Kb + (long)(((1 - m) * BATCH + b) * LS) * DIM;   // kv swap
  const _Float16* Vtp = Vt + (long)((1 - m) * BATCH + b) * DIM * LS;     // [d][s]
  _Float16* AOp = AO + (long)((m * BATCH + b) * LT) * DIM;
  float* AFFp = AFF + (long)(m * BATCH + b) * LS * LT;

  for (int i = tid; i < LS * LT; i += 256) aff[i] = 0.0f;
  __syncthreads();

  for (int h = 0; h < HEADS; ++h) {
    int ch = h * HD;
    { // logits = Q K^T * scale : 4x16 tiles, wave -> (mtile, 8 ntiles)
      int mt = wave & 3, ng = wave >> 2;
      v8f acc[8] = {};
#pragma unroll
      for (int kk = 0; kk < 2; ++kk) {
        v16h a = frag_a(Qp, DIM, 16 * mt, ch + 32 * kk);
#pragma unroll
        for (int i = 0; i < 8; ++i) {
          v16h bf = frag_b(Kp, DIM, 16 * (ng * 8 + i), ch + 32 * kk);
          acc[i] = wmma_f16(a, bf, acc[i]);
        }
      }
      int hl = l >> 4, cl = l & 15;
#pragma unroll
      for (int i = 0; i < 8; ++i)
#pragma unroll
        for (int v = 0; v < 8; ++v)
          logits[(16 * mt + 8 * hl + v) * LS + 16 * (ng * 8 + i) + cl] =
              acc[i][v] * 0.125f;  // 64^-0.5
    }
    __syncthreads();
    // softmax rows (wave handles 8 rows; lane covers 8 strided cols)
    for (int row = wave * 8; row < wave * 8 + 8; ++row) {
      float* Lr = logits + row * LS;
      float vals[8], mx = -1e30f;
#pragma unroll
      for (int i = 0; i < 8; ++i) { vals[i] = Lr[l + 32 * i]; mx = fmaxf(mx, vals[i]); }
#pragma unroll
      for (int o = 16; o > 0; o >>= 1) mx = fmaxf(mx, __shfl_xor(mx, o));
      float sum = 0.0f;
#pragma unroll
      for (int i = 0; i < 8; ++i) { vals[i] = expf(vals[i] - mx); sum += vals[i]; }
#pragma unroll
      for (int o = 16; o > 0; o >>= 1) sum += __shfl_xor(sum, o);
      float inv = 1.0f / sum;
#pragma unroll
      for (int i = 0; i < 8; ++i) Lr[l + 32 * i] = vals[i] * inv;
    }
    __syncthreads();
    // affinity[s][t] += attn[t][s]  (sum over heads)
    for (int i = tid; i < LS * LT; i += 256) {
      int s = i >> 6, t = i & 63;
      aff[i] += logits[t * LS + s];
    }
    // out = attn @ V : 4x4 tiles, 2 per wave; B from transposed V
#pragma unroll
    for (int q = 0; q < 2; ++q) {
      int tau = wave * 2 + q;
      int mt = tau >> 2, nt = tau & 3;
      v8f acc = {};
#pragma unroll
      for (int kk = 0; kk < 8; ++kk) {
        v16h a  = frag_a_f32(logits, LS, 16 * mt, 32 * kk);
        v16h bf = frag_b(Vtp, LS, ch + 16 * nt, 32 * kk);
        acc = wmma_f16(a, bf, acc);
      }
      int hl = l >> 4, cl = l & 15;
#pragma unroll
      for (int v = 0; v < 8; ++v)
        AOp[(long)(16 * mt + 8 * hl + v) * DIM + ch + 16 * nt + cl] = (_Float16)acc[v];
    }
    __syncthreads();  // logits reused next head
  }
  for (int i = tid; i < LS * LT; i += 256) AFFp[i] = aff[i];
}

// ---------- combine: fuse_z = z + crs + lora(other modality); writes Y rows 0..63
// and a transposed f16 copy for the fusion GEMM.
__global__ void combine_kernel(const float* __restrict__ x0, const float* __restrict__ x1,
                               const float* __restrict__ CRS, const float* __restrict__ LUP,
                               float* __restrict__ Y, _Float16* __restrict__ FT) {
  long i = (long)blockIdx.x * 256 + threadIdx.x;     // over 8192*768
  if (i >= (long)ZROWS * DIM) return;
  int d = (int)(i % DIM);
  int r = (int)(i / DIM);             // [m][b][t]
  int m = r >> 12;                    // 4096 rows / modality
  int rr = r & 4095;
  int b = rr >> 6, t = rr & 63;
  float z = (m == 0 ? x0 : x1)[(long)(b * TOK + t) * DIM + d];
  float lora = LUP[(long)((1 - m) * 4096 + rr) * DIM + d];   // modality swap
  float f = z + CRS[i] + lora;
  Y[(long)((m * BATCH + b) * TOK + t) * DIM + d] = f;
  FT[(long)(m * BATCH + b) * DIM * LT + (long)d * LT + t] = (_Float16)f;
}

// ---------- fusion: token_S' = x_S/den + (num/den) @ fuse_z ----------
// dyn LDS: aff 256x64 f32 + coeff 256x64 f16 + invden 256 f32  (~99KB)
__global__ void fusion_kernel(const float* __restrict__ AFF, const _Float16* __restrict__ FT,
                              const float* __restrict__ x0, const float* __restrict__ x1,
                              float* __restrict__ Y) {
  extern __shared__ char smem[];
  float*    aff    = (float*)smem;                           // [256][64]
  _Float16* coeff  = (_Float16*)(smem + LS * LT * 4);        // [256][64]
  float*    invden = (float*)(smem + LS * LT * 4 + LS * LT * 2);
  int b = blockIdx.x, m = blockIdx.y;
  int tid = threadIdx.x, wave = tid >> 5, l = tid & 31;
  const float* AFFp = AFF + (long)(m * BATCH + b) * LS * LT;
  const _Float16* FTp = FT + (long)(m * BATCH + b) * DIM * LT;
  const float* xin = (m == 0 ? x0 : x1) + (long)b * TOK * DIM;
  float* Yp = Y + (long)(m * BATCH + b) * TOK * DIM;

  for (int i = tid; i < LS * LT; i += 256) aff[i] = AFFp[i];
  __syncthreads();
  for (int s = wave; s < LS; s += 8) {    // one wave per row, 2 entries/lane
    float a0 = aff[s * LT + l], a1 = aff[s * LT + l + 32];
    float mx = fmaxf(a0, a1);
#pragma unroll
    for (int o = 16; o > 0; o >>= 1) mx = fmaxf(mx, __shfl_xor(mx, o));
    float n0 = (a0 == mx) ? expf(a0 - 12.0f) : 0.0f;   // exact-match mask
    float n1 = (a1 == mx) ? expf(a1 - 12.0f) : 0.0f;
    float sum = n0 + n1;
#pragma unroll
    for (int o = 16; o > 0; o >>= 1) sum += __shfl_xor(sum, o);
    float inv = 1.0f / (1.0f + sum);
    coeff[s * LT + l]      = (_Float16)(n0 * inv);
    coeff[s * LT + l + 32] = (_Float16)(n1 * inv);
    if (l == 0) invden[s] = inv;
  }
  __syncthreads();
  // (256x64) @ (64x768): 16x48 tiles, 96 per wave, K=64 (2 steps)
  for (int q = 0; q < 96; ++q) {
    int tau = wave * 96 + q;
    int mt = tau / 48, nt = tau % 48;
    v8f acc = {};
#pragma unroll
    for (int kk = 0; kk < 2; ++kk) {
      v16h a  = frag_a((const _Float16*)coeff, LT, 16 * mt, 32 * kk);
      v16h bf = frag_b(FTp, LT, 16 * nt, 32 * kk);
      acc = wmma_f16(a, bf, acc);
    }
    int hl = l >> 4, cl = l & 15;
#pragma unroll
    for (int v = 0; v < 8; ++v) {
      int s = 16 * mt + 8 * hl + v;
      int d = 16 * nt + cl;
      Yp[(long)(LT + s) * DIM + d] = xin[(long)(LT + s) * DIM + d] * invden[s] + acc[v];
    }
  }
}

// ------------------------------- launcher ---------------------------------
extern "C" void kernel_launch(void* const* d_in, const int* in_sizes, int n_in,
                              void* d_out, int out_size, void* d_ws, size_t ws_size,
                              hipStream_t stream) {
  const float* x0 = (const float*)d_in[0];
  const float* x1 = (const float*)d_in[1];
  const float* ln1_w = (const float*)d_in[2];
  const float* ln1_b = (const float*)d_in[3];
  const float* ln2_w = (const float*)d_in[4];
  const float* ln2_b = (const float*)d_in[5];
  const float* wq = (const float*)d_in[6];
  const float* wk = (const float*)d_in[7];
  const float* wv = (const float*)d_in[8];
  const float* wo = (const float*)d_in[9];
  const float* lora_down = (const float*)d_in[10];
  const float* lora_up   = (const float*)d_in[11];
  const float* mlp_down  = (const float*)d_in[12];
  const float* mlp_up    = (const float*)d_in[13];
  float* out = (float*)d_out;

  char* ws = (char*)d_ws;
  size_t off = 0;
  auto alloc = [&](size_t bytes) -> void* {
    void* p = ws + off;
    off += (bytes + 255) & ~(size_t)255;
    return p;
  };
  _Float16* WQT = (_Float16*)alloc((size_t)DIM * DIM * 2);
  _Float16* WKT = (_Float16*)alloc((size_t)DIM * DIM * 2);
  _Float16* WVT = (_Float16*)alloc((size_t)DIM * DIM * 2);
  _Float16* WOT = (_Float16*)alloc((size_t)DIM * DIM * 2);
  _Float16* LDT = (_Float16*)alloc((size_t)64 * DIM * 2);
  _Float16* LUT = (_Float16*)alloc((size_t)DIM * 64 * 2);
  _Float16* MDT = (_Float16*)alloc((size_t)64 * DIM * 2);
  _Float16* MUT = (_Float16*)alloc((size_t)DIM * 64 * 2);
  _Float16* LNZ = (_Float16*)alloc((size_t)ZROWS * DIM * 2);   // contiguous with
  _Float16* LNX = (_Float16*)alloc((size_t)XROWS * DIM * 2);   // ... LNX (ln2 alias)
  _Float16* Qb  = (_Float16*)alloc((size_t)ZROWS * DIM * 2);
  _Float16* Kb  = (_Float16*)alloc((size_t)XROWS * DIM * 2);
  _Float16* Vt  = (_Float16*)alloc((size_t)XROWS * DIM * 2);   // [m][b][768][256]
  _Float16* AO  = (_Float16*)alloc((size_t)ZROWS * DIM * 2);
  float*    AFF = (float*)alloc((size_t)2 * BATCH * LS * LT * 4);
  float*    CRS = (float*)alloc((size_t)ZROWS * DIM * 4);
  _Float16* G1  = (_Float16*)alloc((size_t)ZROWS * 64 * 2);
  float*    LUP = (float*)alloc((size_t)ZROWS * DIM * 4);
  _Float16* FT  = (_Float16*)alloc((size_t)ZROWS * DIM * 2);   // [m][b][768][64]
  float*    Y   = (float*)alloc((size_t)YROWS * DIM * 4);
  _Float16* G2  = (_Float16*)alloc((size_t)YROWS * 64 * 2);
  _Float16* LN2 = LNZ;  // alias: LNZ+LNX region == 40960*768 f16, dead by ln2 time

  dim3 blk(256);
  const int GLDS = 2 * 128 * 32 * 2;   // 16KB double-buffered B tile
  // 1) weight prep (transpose to f16 NxK)
  wprep_kernel<<<(DIM * DIM + 255) / 256, blk, 0, stream>>>(wq, WQT, DIM, DIM);
  wprep_kernel<<<(DIM * DIM + 255) / 256, blk, 0, stream>>>(wk, WKT, DIM, DIM);
  wprep_kernel<<<(DIM * DIM + 255) / 256, blk, 0, stream>>>(wv, WVT, DIM, DIM);
  wprep_kernel<<<(DIM * DIM + 255) / 256, blk, 0, stream>>>(wo, WOT, DIM, DIM);
  wprep_kernel<<<(DIM * 64 + 255) / 256, blk, 0, stream>>>(lora_down, LDT, DIM, 64);
  wprep_kernel<<<(64 * DIM + 255) / 256, blk, 0, stream>>>(lora_up, LUT, 64, DIM);
  wprep_kernel<<<(DIM * 64 + 255) / 256, blk, 0, stream>>>(mlp_down, MDT, DIM, 64);
  wprep_kernel<<<(64 * DIM + 255) / 256, blk, 0, stream>>>(mlp_up, MUT, 64, DIM);
  // 2) ln1
  ln1_kernel<<<YROWS, blk, 0, stream>>>(x0, x1, ln1_w, ln1_b, LNZ, LNX);
  // 3) projections (wide, async-LDS staged weights)
  gemm_wide_kernel<GM_F16><<<dim3(ZROWS / 64, 6), blk, GLDS, stream>>>(
      LNZ, WQT, ZROWS, DIM, DIM, nullptr, Qb, nullptr, 0);
  gemm_wide_kernel<GM_F16><<<dim3(XROWS / 64, 6), blk, GLDS, stream>>>(
      LNX, WKT, XROWS, DIM, DIM, nullptr, Kb, nullptr, 0);
  gemm_wide_kernel<GM_TRANS_F16><<<dim3(XROWS / 64, 6), blk, GLDS, stream>>>(
      LNX, WVT, XROWS, DIM, DIM, nullptr, Vt, nullptr, LS);
  gemm_narrow_kernel<GM_GELU_F16><<<dim3(ZROWS / 128, 1), blk, 0, stream>>>(
      LNZ, LDT, ZROWS, 64, DIM, nullptr, G1, nullptr, 0);
  // 4) attention (128KB dynamic LDS; WGP has 320KB)
  attn_kernel<<<dim3(BATCH, 2), blk, 2 * 64 * 256 * 4, stream>>>(Qb, Kb, Vt, AO, AFF);
  // 5) output proj + lora-up + combine
  gemm_wide_kernel<GM_F32><<<dim3(ZROWS / 64, 6), blk, GLDS, stream>>>(
      AO, WOT, ZROWS, DIM, DIM, CRS, nullptr, nullptr, 0);
  gemm_wide_kernel<GM_F32><<<dim3(ZROWS / 64, 6), blk, GLDS, stream>>>(
      G1, LUT, ZROWS, DIM, 64, LUP, nullptr, nullptr, 0);
  combine_kernel<<<(int)(((long)ZROWS * DIM + 255) / 256), blk, 0, stream>>>(
      x0, x1, CRS, LUP, Y, FT);
  // 6) fusion (writes Y rows 64..319)
  fusion_kernel<<<dim3(BATCH, 2), blk, LS * LT * 4 + LS * LT * 2 + LS * 4, stream>>>(
      AFF, FT, x0, x1, Y);
  // 7) ln2 + MLP lora, residual-add straight into d_out
  ln2_kernel<<<YROWS, blk, 0, stream>>>(Y, ln2_w, ln2_b, LN2);
  gemm_narrow_kernel<GM_GELU_F16><<<dim3(YROWS / 128, 1), blk, 0, stream>>>(
      LN2, MDT, YROWS, 64, DIM, nullptr, G2, nullptr, 0);
  gemm_wide_kernel<GM_RESID_F32><<<dim3(YROWS / 64, 6), blk, GLDS, stream>>>(
      G2, MUT, YROWS, DIM, 64, out, nullptr, Y, 0);
}
